// GraphAttention_Naive_59476707115119
// MI455X (gfx1250) — matrix operand order repaired and checked by reading, not compiled, not run on hardware
//
#include <hip/hip_runtime.h>

typedef _Float16 half_t;
typedef __attribute__((ext_vector_type(16))) _Float16 v16h;
typedef __attribute__((ext_vector_type(8)))  float    v8f;

#define D_MODEL 1024
#define N_HEADS 16
#define D_HEAD  64
#define BATCH   4
#define SEQ     2048
#define ROWS    (BATCH * SEQ)   /* 8192 */

// ---------------------------------------------------------------------------
// WMMA fragment loaders (contiguous k order k = 16*(lane/16) + elem, wave32)
// ---------------------------------------------------------------------------

// A fragment: 16x32 (MxK), source row-major with leading dim `ld`.
// lane<16: M=lane, elems 0..15 = K0..15
// lane>=16: M=lane-16, elems 0..15 = K16..31
__device__ __forceinline__ v16h load_A16x32(const half_t* __restrict__ p, int ld,
                                            int row0, int k0) {
  int lane = threadIdx.x & 31;
  int m    = lane & 15;
  int kb   = k0 + ((lane & 16) ? 16 : 0);
  const half_t* r = p + (size_t)(row0 + m) * ld + kb;
  v16h a;
#pragma unroll
  for (int i = 0; i < 8; ++i) a[i] = r[i];
#pragma unroll
  for (int i = 0; i < 8; ++i) a[8 + i] = r[8 + i];
  return a;
}

// B fragment: 32x16 (KxN) where memory is laid out [n][k] (K contiguous per N
// row) -- i.e. X * W^T with W row-major, Q * K^T with K row-major, or P * V
// with V pre-transposed in LDS.
// lane<16: N=lane, elems = K0..15 ; lane>=16: N=lane-16, elems = K16..31
__device__ __forceinline__ v16h load_B_kcontig(const half_t* __restrict__ p, int ld,
                                               int n0, int k0) {
  int lane = threadIdx.x & 31;
  int n    = lane & 15;
  int kb   = k0 + ((lane & 16) ? 16 : 0);
  const half_t* r = p + (size_t)(n0 + n) * ld + kb;
  v16h b;
#pragma unroll
  for (int i = 0; i < 16; ++i) b[i] = r[i];
  return b;
}

__device__ __forceinline__ v8f wmma16(v16h a, v16h b, v8f c) {
  return __builtin_amdgcn_wmma_f32_16x16x32_f16(false, a, false, b, (short)0, c,
                                                false, false);
}

// ---------------------------------------------------------------------------
// f32 -> f16 convert
// ---------------------------------------------------------------------------
__global__ __launch_bounds__(256) void cvt_f32_f16_kernel(
    const float* __restrict__ src, half_t* __restrict__ dst, int n) {
  int i = blockIdx.x * blockDim.x + threadIdx.x;
  int stride = gridDim.x * blockDim.x;
  for (; i < n; i += stride) dst[i] = (half_t)src[i];
}

// ---------------------------------------------------------------------------
// Fused QKV projection:  Y = X * W^T + b, stored head-major [proj][B*H][N][64]
// Block: 256 threads (8 waves). Wave tile: 32 rows x 64 cols (2 A frags share
// each B frag -> 8 WMMA per 6 fragment loads).
// Grid: (ROWS/256, D_MODEL/64, 3)
// ---------------------------------------------------------------------------
__global__ __launch_bounds__(256) void qkv_proj_kernel(
    const half_t* __restrict__ X,    // [ROWS][D_MODEL]
    const half_t* __restrict__ Wh,   // [3][D_MODEL][D_MODEL]
    const float* __restrict__ bq, const float* __restrict__ bk,
    const float* __restrict__ bv,
    half_t* __restrict__ QKV)        // [3][B*H][SEQ][D_HEAD]
{
  int proj = blockIdx.z;
  const half_t* W    = Wh + (size_t)proj * D_MODEL * D_MODEL;
  const float*  bias = (proj == 0) ? bq : ((proj == 1) ? bk : bv);
  half_t*       dst  = QKV + (size_t)proj * ROWS * D_MODEL;

  int wave = threadIdx.x >> 5;
  int lane = threadIdx.x & 31;
  int row0 = blockIdx.x * 256 + wave * 32;
  int col0 = blockIdx.y * 64;

  v8f acc[2][4];
#pragma unroll
  for (int rt = 0; rt < 2; ++rt)
#pragma unroll
    for (int t = 0; t < 4; ++t)
#pragma unroll
      for (int i = 0; i < 8; ++i) acc[rt][t][i] = 0.f;

  for (int kt = 0; kt < D_MODEL; kt += 32) {
    v16h a0 = load_A16x32(X, D_MODEL, row0, kt);
    v16h a1 = load_A16x32(X, D_MODEL, row0 + 16, kt);
#pragma unroll
    for (int t = 0; t < 4; ++t) {
      v16h b = load_B_kcontig(W, D_MODEL, col0 + t * 16, kt);
      acc[0][t] = wmma16(a0, b, acc[0][t]);
      acc[1][t] = wmma16(a1, b, acc[1][t]);
    }
  }

  int hi = (lane & 16) ? 8 : 0;
  int cn = lane & 15;
#pragma unroll
  for (int t = 0; t < 4; ++t) {
    int col = col0 + t * 16 + cn;
    int h = col >> 6, d = col & 63;
    float bb = bias[col];
#pragma unroll
    for (int rt = 0; rt < 2; ++rt) {
#pragma unroll
      for (int i = 0; i < 8; ++i) {
        int row = row0 + rt * 16 + i + hi;
        int b = row >> 11, n = row & (SEQ - 1);
        dst[(((size_t)(b * N_HEADS + h) * SEQ) + n) * D_HEAD + d] =
            (half_t)(acc[rt][t][i] + bb);
      }
    }
  }
}

// ---------------------------------------------------------------------------
// Flash attention. Block: 128 threads (4 waves); each wave owns 16 query rows.
// K tile staged via async global->LDS DMA (ASYNCcnt path); V tile staged
// transposed through VGPRs (+2 half pad). All WMMA fragments read contiguous
// LDS. Grid: (SEQ/64, BATCH*N_HEADS)
// ---------------------------------------------------------------------------
__global__ __launch_bounds__(128) void flash_attn_kernel(
    const half_t* __restrict__ QKV,          // [3][B*H][SEQ][D_HEAD] f16
    const unsigned char* __restrict__ mask,  // [B][SEQ] bool
    half_t* __restrict__ Ymid)               // [ROWS][D_MODEL] f16
{
  __shared__ half_t Ks[32 * 64];        // K tile, raw [key][d]
  __shared__ half_t Vt[64 * 34];        // V tile, transposed [d][key] (+2 pad)
  __shared__ half_t Plds[4][16 * 32];   // per-wave P transpose buffer

  int wave = threadIdx.x >> 5;
  int lane = threadIdx.x & 31;
  int bh   = blockIdx.y;
  int b = bh >> 4, h = bh & 15;

  const size_t headoff = (size_t)bh * SEQ * D_HEAD;
  const half_t* Q = QKV + headoff;
  const half_t* K = QKV + (size_t)1 * ROWS * D_MODEL + headoff;
  const half_t* V = QKV + (size_t)2 * ROWS * D_MODEL + headoff;
  const unsigned char* mrow = mask + (size_t)b * SEQ;

  int qrow0 = blockIdx.x * 64 + wave * 16;

  // Q fragments for this wave's 16 rows (d = 0..31, 32..63), kept in VGPRs.
  v16h aq0 = load_A16x32(Q, D_HEAD, qrow0, 0);
  v16h aq1 = load_A16x32(Q, D_HEAD, qrow0, 32);

  float m[8], l[8], alpha[8];
  v8f acc[4];
#pragma unroll
  for (int i = 0; i < 8; ++i) { m[i] = -1e30f; l[i] = 0.f; }
#pragma unroll
  for (int t = 0; t < 4; ++t)
#pragma unroll
    for (int i = 0; i < 8; ++i) acc[t][i] = 0.f;

  const float kscale = 0.125f;  // 1/sqrt(64)
  int hi = (lane & 16) ? 8 : 0;
  int cn = lane & 15;
  half_t* pl = &Plds[wave][0];

  // cooperative-staging coordinates: thread -> (key row, 16-wide d slice)
  int tkey = threadIdx.x >> 2;        // 0..31
  int td   = (threadIdx.x & 3) * 16;  // 0,16,32,48

  // LDS byte address of this thread's K-tile destination (generic LDS
  // pointers carry the LDS offset in their low 32 bits, ISA 10.2).
  unsigned kdst = (unsigned)(size_t)&Ks[tkey * 64 + td];

  for (int j = 0; j < SEQ; j += 32) {
    // ---- K tile: async global->LDS DMA, 32 B per thread (no VGPR transit).
    // The instruction offset applies to both LDS and global addresses.
    {
      unsigned long long ksrc = (unsigned long long)(K + (size_t)(j + tkey) * D_HEAD + td);
      asm volatile(
          "global_load_async_to_lds_b128 %0, %1, off\n\t"
          "global_load_async_to_lds_b128 %0, %1, off offset:16"
          :
          : "v"(kdst), "v"(ksrc)
          : "memory");
    }
    // ---- V tile: transposed store through VGPRs (overlaps with async K) ----
    const half_t* vsrc = V + (size_t)(j + tkey) * D_HEAD + td;
#pragma unroll
    for (int i = 0; i < 16; ++i) Vt[(td + i) * 34 + tkey] = vsrc[i];
    asm volatile("s_wait_asynccnt 0x0" ::: "memory");
    __syncthreads();

    // ---- S = Q * K^T for 32 keys (two 16-key N tiles, two 32-d K steps) ----
    v16h bk0a = load_B_kcontig(Ks, 64, 0, 0);
    v16h bk0b = load_B_kcontig(Ks, 64, 0, 32);
    v16h bk1a = load_B_kcontig(Ks, 64, 16, 0);
    v16h bk1b = load_B_kcontig(Ks, 64, 16, 32);

    v8f s0, s1;
#pragma unroll
    for (int i = 0; i < 8; ++i) { s0[i] = 0.f; s1[i] = 0.f; }
    s0 = wmma16(aq0, bk0a, s0);
    s0 = wmma16(aq1, bk0b, s0);
    s1 = wmma16(aq0, bk1a, s1);
    s1 = wmma16(aq1, bk1b, s1);

    bool ok0 = mrow[j + cn] != 0;
    bool ok1 = mrow[j + 16 + cn] != 0;

    // ---- online softmax (per-row across the 16-lane C fragment groups) ----
#pragma unroll
    for (int i = 0; i < 8; ++i) {
      float v0 = ok0 ? s0[i] * kscale : -1e30f;
      float v1 = ok1 ? s1[i] * kscale : -1e30f;
      float tmax = fmaxf(v0, v1);
#pragma unroll
      for (int off = 1; off < 16; off <<= 1)
        tmax = fmaxf(tmax, __shfl_xor(tmax, off, 32));
      float mnew = fmaxf(m[i], tmax);
      float al = __expf(m[i] - mnew);
      float p0 = __expf(v0 - mnew);
      float p1 = __expf(v1 - mnew);
      float rs = p0 + p1;
#pragma unroll
      for (int off = 1; off < 16; off <<= 1) rs += __shfl_xor(rs, off, 32);
      l[i] = l[i] * al + rs;
      m[i] = mnew;
      alpha[i] = al;
      s0[i] = p0;
      s1[i] = p1;
    }

    // rescale running output accumulators
#pragma unroll
    for (int t = 0; t < 4; ++t)
#pragma unroll
      for (int i = 0; i < 8; ++i) acc[t][i] *= alpha[i];

    // ---- transpose P (C layout) -> A layout through per-wave LDS tile ----
#pragma unroll
    for (int i = 0; i < 8; ++i) {
      int r = i + hi;
      pl[r * 32 + cn]      = (half_t)s0[i];
      pl[r * 32 + 16 + cn] = (half_t)s1[i];
    }
    asm volatile("s_wait_dscnt 0x0" ::: "memory");
    v16h ap = load_A16x32(pl, 32, 0, 0);

    // ---- O += P * V (four 16-col N tiles over d_head = 64) ----
#pragma unroll
    for (int t = 0; t < 4; ++t) {
      v16h bv16 = load_B_kcontig(Vt, 34, t * 16, 0);
      acc[t] = wmma16(ap, bv16, acc[t]);
    }
    __syncthreads();  // protect Ks/Vt before next iteration's staging
  }

  // ---- normalize and write back in [B*N][H*64] layout ----
#pragma unroll
  for (int t = 0; t < 4; ++t) {
#pragma unroll
    for (int i = 0; i < 8; ++i) {
      float o = acc[t][i] / l[i];
      int r   = qrow0 + i + hi;
      int col = h * 64 + t * 16 + cn;
      Ymid[(size_t)(b * SEQ + r) * D_MODEL + col] = (half_t)o;
    }
  }
}

// ---------------------------------------------------------------------------
// Output projection: out = Y * Wo^T (f32 out). Same 32x64 wave tiling.
// Grid: (ROWS/256, D_MODEL/64)
// ---------------------------------------------------------------------------
__global__ __launch_bounds__(256) void out_proj_kernel(
    const half_t* __restrict__ Y,   // [ROWS][D_MODEL]
    const half_t* __restrict__ Wo,  // [D_MODEL][D_MODEL]
    float* __restrict__ out)        // [ROWS][D_MODEL]
{
  int wave = threadIdx.x >> 5;
  int lane = threadIdx.x & 31;
  int row0 = blockIdx.x * 256 + wave * 32;
  int col0 = blockIdx.y * 64;

  v8f acc[2][4];
#pragma unroll
  for (int rt = 0; rt < 2; ++rt)
#pragma unroll
    for (int t = 0; t < 4; ++t)
#pragma unroll
      for (int i = 0; i < 8; ++i) acc[rt][t][i] = 0.f;

  for (int kt = 0; kt < D_MODEL; kt += 32) {
    v16h a0 = load_A16x32(Y, D_MODEL, row0, kt);
    v16h a1 = load_A16x32(Y, D_MODEL, row0 + 16, kt);
#pragma unroll
    for (int t = 0; t < 4; ++t) {
      v16h b = load_B_kcontig(Wo, D_MODEL, col0 + t * 16, kt);
      acc[0][t] = wmma16(a0, b, acc[0][t]);
      acc[1][t] = wmma16(a1, b, acc[1][t]);
    }
  }

  int hi = (lane & 16) ? 8 : 0;
  int cn = lane & 15;
#pragma unroll
  for (int t = 0; t < 4; ++t) {
    int col = col0 + t * 16 + cn;
#pragma unroll
    for (int rt = 0; rt < 2; ++rt) {
#pragma unroll
      for (int i = 0; i < 8; ++i) {
        int row = row0 + rt * 16 + i + hi;
        out[(size_t)row * D_MODEL + col] = acc[rt][t][i];
      }
    }
  }
}

// ---------------------------------------------------------------------------
extern "C" void kernel_launch(void* const* d_in, const int* in_sizes, int n_in,
                              void* d_out, int out_size, void* d_ws, size_t ws_size,
                              hipStream_t stream) {
  const float*         x    = (const float*)d_in[0];
  const unsigned char* mask = (const unsigned char*)d_in[1];
  const float*         Wq   = (const float*)d_in[2];
  const float*         bq   = (const float*)d_in[3];
  const float*         Wk   = (const float*)d_in[4];
  const float*         bk   = (const float*)d_in[5];
  const float*         Wv   = (const float*)d_in[6];
  const float*         bv   = (const float*)d_in[7];
  const float*         Wo   = (const float*)d_in[8];
  float*               out  = (float*)d_out;

  // Workspace carve-up (total ~92 MB)
  char* ws = (char*)d_ws;
  half_t* Xh   = (half_t*)ws;  ws += (size_t)ROWS * D_MODEL * sizeof(half_t);
  half_t* Wh   = (half_t*)ws;  ws += (size_t)3 * D_MODEL * D_MODEL * sizeof(half_t);
  half_t* Woh  = (half_t*)ws;  ws += (size_t)D_MODEL * D_MODEL * sizeof(half_t);
  half_t* QKV  = (half_t*)ws;  ws += (size_t)3 * ROWS * D_MODEL * sizeof(half_t);
  half_t* Ymid = (half_t*)ws;  ws += (size_t)ROWS * D_MODEL * sizeof(half_t);

  const int nW = D_MODEL * D_MODEL;
  cvt_f32_f16_kernel<<<1024, 256, 0, stream>>>(x, Xh, ROWS * D_MODEL);
  cvt_f32_f16_kernel<<<256, 256, 0, stream>>>(Wq, Wh, nW);
  cvt_f32_f16_kernel<<<256, 256, 0, stream>>>(Wk, Wh + (size_t)nW, nW);
  cvt_f32_f16_kernel<<<256, 256, 0, stream>>>(Wv, Wh + (size_t)2 * nW, nW);
  cvt_f32_f16_kernel<<<256, 256, 0, stream>>>(Wo, Woh, nW);

  dim3 g1(ROWS / 256, D_MODEL / 64, 3);
  qkv_proj_kernel<<<g1, 256, 0, stream>>>(Xh, Wh, bq, bk, bv, QKV);

  dim3 g2(SEQ / 64, BATCH * N_HEADS);
  flash_attn_kernel<<<g2, 128, 0, stream>>>(QKV, mask, Ymid);

  dim3 g3(ROWS / 256, D_MODEL / 64);
  out_proj_kernel<<<g3, 256, 0, stream>>>(Ymid, Woh, out);
}
